// StaticGCN_7885559955671
// MI455X (gfx1250) — compile-verified, matrix operand order checked by reading, not run on hardware
//
#include <hip/hip_runtime.h>

#define NN 10000      // num nodes
#define NE 640000     // num edges
#define CH 128        // IN = HID = OUT channels

typedef __attribute__((ext_vector_type(16))) _Float16 v16h;
typedef __attribute__((ext_vector_type(8)))  _Float16 v8h;
typedef __attribute__((ext_vector_type(8)))  float    v8f;

// ---------------- degree / normalization ----------------

__global__ void k_init_deg(float* __restrict__ deg) {
    int i = blockIdx.x * blockDim.x + threadIdx.x;
    if (i < NN) deg[i] = 1.0f;                 // self-loop weight
}

__global__ void k_deg_edges(const int* __restrict__ ei, const float* __restrict__ w,
                            float* __restrict__ deg) {
    int e = blockIdx.x * blockDim.x + threadIdx.x;
    if (e < NE) atomicAdd(&deg[ei[NE + e]], w[e]);   // segment_sum over col
}

__global__ void k_dinv(float* __restrict__ deg) {
    int i = blockIdx.x * blockDim.x + threadIdx.x;
    if (i < NN) {
        float d = deg[i];
        deg[i] = (d > 0.0f) ? rsqrtf(d) : 0.0f;      // in-place deg -> deg^{-1/2}
    }
}

// ---------------- conversions ----------------

__global__ void k_fill_ones_h(_Float16* __restrict__ x, int n) {
    int i = blockIdx.x * blockDim.x + threadIdx.x;
    if (i < n) x[i] = (_Float16)1.0f;
}

__global__ void k_f32_to_f16(const float* __restrict__ s, _Float16* __restrict__ d, int n) {
    int i = blockIdx.x * blockDim.x + threadIdx.x;
    if (i < n) d[i] = (_Float16)s[i];
}

// W (CH x CH, row major, f32) -> Wt (CH x CH, f16) with Wt[n*CH+k] = W[k*CH+n]
__global__ void k_transpose_f16(const float* __restrict__ W, _Float16* __restrict__ Wt) {
    int idx = blockIdx.x * blockDim.x + threadIdx.x;
    if (idx < CH * CH) {
        int k = idx / CH, n = idx % CH;
        Wt[n * CH + k] = (_Float16)W[idx];
    }
}

__global__ void k_zero(float* __restrict__ p, int n) {
    int i = blockIdx.x * blockDim.x + threadIdx.x;
    if (i < n) p[i] = 0.0f;
}

// ---------------- WMMA GEMM: C[M x 128] = A[M x 128] * W[128 x 128] ----------------
// A: row-major f16.  Bt: f16, Bt[n*128+k] = W[k][n] (column of W contiguous).
// One wave per 16x16 output tile; K=128 consumed as 4x v_wmma_f32_16x16x32_f16.
__global__ void k_gemm_wmma(const _Float16* __restrict__ A,
                            const _Float16* __restrict__ Bt,
                            float* __restrict__ Cm) {
    int wave = (int)((blockIdx.x * blockDim.x + threadIdx.x) >> 5);
    int lane = threadIdx.x & 31;
    int tileM = wave >> 3;        // 625 tiles along M
    int tileN = wave & 7;         // 8 tiles along N
    if (tileM >= NN / 16) return;

    int mrow = tileM * 16 + (lane & 15);   // A row this lane supplies
    int nrow = tileN * 16 + (lane & 15);   // W column this lane supplies
    int kb   = (lane & 16) ? 8 : 0;        // K phase per ISA 16-bit A layout

    v8f acc = {};
#pragma unroll
    for (int kt = 0; kt < CH; kt += 32) {
        const _Float16* ap = A  + (size_t)mrow * CH + kt + kb;
        const _Float16* bp = Bt + (size_t)nrow * CH + kt + kb;
        // halves 0..7 -> K = kb+0..7 ; halves 8..15 -> K = kb+16..23
        v8h a0 = *(const v8h*)(ap);
        v8h a1 = *(const v8h*)(ap + 16);
        v8h b0 = *(const v8h*)(bp);
        v8h b1 = *(const v8h*)(bp + 16);
        v16h av, bv;
#pragma unroll
        for (int i = 0; i < 8; ++i) {
            av[i] = a0[i]; av[i + 8] = a1[i];
            bv[i] = b0[i]; bv[i + 8] = b1[i];
        }
        acc = __builtin_amdgcn_wmma_f32_16x16x32_f16(
            /*neg_a=*/false, av, /*neg_b=*/false, bv,
            /*c_mod=*/(short)0, acc, /*reuse_a=*/false, /*reuse_b=*/false);
    }

    // C/D layout: lanes 0-15 -> M=r, lanes 16-31 -> M=r+8; N = lane%16
    int ncol = tileN * 16 + (lane & 15);
    int mbase = tileM * 16 + ((lane & 16) ? 8 : 0);
#pragma unroll
    for (int r = 0; r < 8; ++r)
        Cm[(size_t)(mbase + r) * CH + ncol] = acc[r];
}

// ---------------- edge scatter: agg[col] += dinv[row]*w*dinv[col] * xw[row] ----------------
// one wave32 per edge, each lane owns 4 channels (float4 gather, 4 atomics)
__global__ void k_scatter(const int* __restrict__ ei, const float* __restrict__ w,
                          const float* __restrict__ dinv, const float* __restrict__ xw,
                          float* __restrict__ agg) {
    int wave = (int)((blockIdx.x * blockDim.x + threadIdx.x) >> 5);
    int lane = threadIdx.x & 31;
    if (wave >= NE) return;
    int row = ei[wave];
    int col = ei[NE + wave];
    float norm = dinv[row] * w[wave] * dinv[col];
    const float4 x = *(const float4*)(xw + (size_t)row * CH + lane * 4);
    float* o = agg + (size_t)col * CH + lane * 4;
    atomicAdd(o + 0, norm * x.x);
    atomicAdd(o + 1, norm * x.y);
    atomicAdd(o + 2, norm * x.z);
    atomicAdd(o + 3, norm * x.w);
}

// self-loop term + bias (+ optional relu)
__global__ void k_finalize(const float* __restrict__ agg, const float* __restrict__ xw,
                           const float* __restrict__ dinv, const float* __restrict__ bias,
                           float* __restrict__ out, int relu) {
    int idx = blockIdx.x * blockDim.x + threadIdx.x;
    if (idx >= NN * CH) return;
    int c = idx >> 7, j = idx & (CH - 1);
    float di = dinv[c];
    float v = agg[idx] + di * di * xw[idx] + bias[j];
    if (relu) v = fmaxf(v, 0.0f);
    out[idx] = v;
}

// global mean pool: out[j] = mean_c h[c][j]; one block per channel
__global__ void k_mean(const float* __restrict__ h, float* __restrict__ out) {
    __shared__ float s[256];
    int j = blockIdx.x;
    float acc = 0.0f;
    for (int c = threadIdx.x; c < NN; c += blockDim.x)
        acc += h[(size_t)c * CH + j];
    s[threadIdx.x] = acc;
    __syncthreads();
    for (int st = 128; st > 0; st >>= 1) {
        if ((int)threadIdx.x < st) s[threadIdx.x] += s[threadIdx.x + st];
        __syncthreads();
    }
    if (threadIdx.x == 0) out[j] = s[0] / (float)NN;
}

// ---------------- orchestration ----------------

extern "C" void kernel_launch(void* const* d_in, const int* in_sizes, int n_in,
                              void* d_out, int out_size, void* d_ws, size_t ws_size,
                              hipStream_t stream) {
    (void)in_sizes; (void)n_in; (void)out_size; (void)ws_size;
    const int*   ei = (const int*)d_in[1];     // [2, E]
    const float* ew = (const float*)d_in[2];   // [E]
    const float* W1 = (const float*)d_in[3];
    const float* b1 = (const float*)d_in[4];
    const float* W2 = (const float*)d_in[5];
    const float* b2 = (const float*)d_in[6];
    float* out = (float*)d_out;

    // carve workspace (256B aligned)
    char* ws = (char*)d_ws;
    size_t off = 0;
    auto carve = [&](size_t bytes) -> void* {
        void* p = ws + off;
        off += (bytes + 255) & ~(size_t)255;
        return p;
    };
    float*    dinv = (float*)   carve((size_t)NN * 4);
    _Float16* Xh   = (_Float16*)carve((size_t)NN * CH * 2);
    _Float16* Wh   = (_Float16*)carve((size_t)CH * CH * 2);
    float*    xw   = (float*)   carve((size_t)NN * CH * 4);
    float*    agg  = (float*)   carve((size_t)NN * CH * 4);
    float*    h    = (float*)   carve((size_t)NN * CH * 4);

    const int T = 256;
    const int gN   = (NN + T - 1) / T;
    const int gE   = (NE + T - 1) / T;
    const int gNC  = (NN * CH + T - 1) / T;
    const int gWW  = (CH * CH + T - 1) / T;
    const int gGEMM = (NN / 16) * 8 / 8;       // 5000 waves / 8 waves per block
    const int gSCAT = NE / 8;                  // 8 edges (waves) per 256-thread block

    // normalization: deg -> deg^{-1/2}  (dinv buffer holds deg first)
    k_init_deg <<<gN, T, 0, stream>>>(dinv);
    k_deg_edges<<<gE, T, 0, stream>>>(ei, ew, dinv);
    k_dinv     <<<gN, T, 0, stream>>>(dinv);

    // ---- layer 1: X = ones ----
    k_fill_ones_h <<<gNC, T, 0, stream>>>(Xh, NN * CH);
    k_transpose_f16<<<gWW, T, 0, stream>>>(W1, Wh);
    k_gemm_wmma   <<<gGEMM, T, 0, stream>>>(Xh, Wh, xw);
    k_zero        <<<gNC, T, 0, stream>>>(agg, NN * CH);
    k_scatter     <<<gSCAT, T, 0, stream>>>(ei, ew, dinv, xw, agg);
    k_finalize    <<<gNC, T, 0, stream>>>(agg, xw, dinv, b1, h, /*relu=*/1);

    // ---- layer 2: X = h ----
    k_f32_to_f16  <<<gNC, T, 0, stream>>>(h, Xh, NN * CH);
    k_transpose_f16<<<gWW, T, 0, stream>>>(W2, Wh);
    k_gemm_wmma   <<<gGEMM, T, 0, stream>>>(Xh, Wh, xw);
    k_zero        <<<gNC, T, 0, stream>>>(agg, NN * CH);
    k_scatter     <<<gSCAT, T, 0, stream>>>(ei, ew, dinv, xw, agg);
    k_finalize    <<<gNC, T, 0, stream>>>(agg, xw, dinv, b2, h, /*relu=*/0);

    // ---- mean pool -> [1, 128] ----
    k_mean<<<CH, T, 0, stream>>>(h, out);
}
